// Seq2Seq_82669530513597
// MI455X (gfx1250) — compile-verified, matrix operand order checked by reading, not run on hardware
//
#include <hip/hip_runtime.h>
#include <hip/hip_bf16.h>

// ---------------------------------------------------------------------------
// Seq2Seq GRU encoder/decoder for MI455X (gfx1250, wave32, WMMA).
// Heavy math (output projection 3136x32000x636 and h@W_hh^T recurrences) runs
// on v_wmma_f32_16x16x32_bf16 with f32 accumulation.
// ---------------------------------------------------------------------------

typedef __attribute__((ext_vector_type(16))) __bf16 v16bf;
typedef __attribute__((ext_vector_type(8)))  __bf16 v8bf;
typedef __attribute__((ext_vector_type(8)))  float  v8f;

union FragBF { v16bf v; v8bf h[2]; };

#define WMMA_BF16(a, b, c) \
  __builtin_amdgcn_wmma_f32_16x16x32_bf16(false, (a), false, (b), (short)0, (c), false, false)

// A/B fragment (16x32 / 32x16, bf16) layout per CDNA5 ISA 7.12.2:
// lanes 0-15: row = lane, K = kb+{0..7} and kb+16+{0..7}
// lanes 16-31: row = lane-16, K = kb+8+{0..7} and kb+24+{0..7}
// => two contiguous 16-byte loads per lane.
static __device__ __forceinline__ v16bf frag_ld(const __bf16* rowptr, int kb, int lane) {
  const int half = (lane >> 4) & 1;
  const __bf16* p = rowptr + kb + half * 8;
  FragBF f;
  f.h[0] = *reinterpret_cast<const v8bf*>(p);
  f.h[1] = *reinterpret_cast<const v8bf*>(p + 16);
  return f.v;
}

// ------------------------------ constants ---------------------------------
#define EMB   128
#define HID   254
#define G3    762      // 3*H
#define VOC   32000
#define BAT   64
#define KPAD  256      // padded K for recurrence (H=254 -> 256)
#define FK    640      // padded K for output GEMM (636 -> 640)
#define LDH   264      // bf16 elements per h row in LDS (256 + 8 skew)
#define LDG   768      // f32 elements per gh row in LDS
#define RB    16       // batch rows per recurrence block

// --------------------------- trivial kernels ------------------------------

__global__ void zero_f32_kernel(float* __restrict__ p, int n) {
  int i = blockIdx.x * blockDim.x + threadIdx.x;
  if (i < n) p[i] = 0.0f;
}

// dst[m][k] = emb[tok[m]][k], k<128
__global__ void gather_kernel(const int* __restrict__ tok, const float* __restrict__ emb,
                              float* __restrict__ dst, int nrows) {
  int idx = blockIdx.x * blockDim.x + threadIdx.x;
  if (idx >= nrows * EMB) return;
  int m = idx >> 7, k = idx & 127;
  dst[idx] = emb[(size_t)tok[m] * EMB + k];
}

// f32 -> bf16 with zero padding to (rows_out x kout)
__global__ void conv_bf16_kernel(const float* __restrict__ src, __bf16* __restrict__ dst,
                                 int rows_in, int kin, int rows_out, int kout) {
  int idx = blockIdx.x * blockDim.x + threadIdx.x;
  if (idx >= rows_out * kout) return;
  int r = idx / kout, k = idx - r * kout;
  float v = (r < rows_in && k < kin) ? src[(size_t)r * kin + k] : 0.0f;
  dst[idx] = (__bf16)v;
}

// ----------------------- gi precompute (parallel) --------------------------

// gi[m][n] = emb[m][:]·W_ih[n][:] + b_ih[n], K=128
__global__ void gi_enc_kernel(const float* __restrict__ emb, const float* __restrict__ wih,
                              const float* __restrict__ bih, float* __restrict__ gi, int M) {
  int idx = blockIdx.x * blockDim.x + threadIdx.x;
  if (idx >= M * G3) return;
  int m = idx / G3, n = idx - m * G3;
  const float4* a = reinterpret_cast<const float4*>(emb + (size_t)m * EMB);
  const float4* w = reinterpret_cast<const float4*>(wih + (size_t)n * EMB);
  float s = bih[n];
#pragma unroll 8
  for (int k = 0; k < EMB / 4; ++k) {
    float4 av = a[k], wv = w[k];
    s += av.x * wv.x + av.y * wv.y + av.z * wv.z + av.w * wv.w;
  }
  gi[idx] = s;
}

// gi[m][n] = [dec_emb[m], context[b]]·W_ih_d[n][:] + b_ih_d[n], K=382
__global__ void gi_dec_kernel(const float* __restrict__ demb, const float* __restrict__ ctx,
                              const float* __restrict__ wih, const float* __restrict__ bih,
                              float* __restrict__ gi) {
  int idx = blockIdx.x * blockDim.x + threadIdx.x;
  if (idx >= (49 * BAT) * G3) return;
  int m = idx / G3, n = idx - m * G3;
  int b = m & 63;
  const float* e = demb + (size_t)m * EMB;
  const float* c = ctx + (size_t)b * HID;
  const float* w = wih + (size_t)n * (EMB + HID);
  float s = bih[n];
#pragma unroll 4
  for (int k = 0; k < EMB; ++k) s += e[k] * w[k];
#pragma unroll 2
  for (int k = 0; k < HID; ++k) s += c[k] * w[EMB + k];
  gi[idx] = s;
}

// -------------------- GRU recurrence (persistent, WMMA) --------------------
// Grid = 4 blocks, each owns 16 batch rows (batches are independent).
// Per step: gh = h @ W_hh^T via v_wmma (16x768 tile, K=256), then fp32 gates.
// A (h) fragments are loaded once per step and reused across all 48 N-tiles;
// B (W_hh) fragments are transient (re-read from L2-resident global) so the
// compiler cannot hoist them across the step loop and spill to scratch.
__global__ __launch_bounds__(512)
void gru_seq_kernel(const float* __restrict__ gi,     // [steps*64][762]
                    const __bf16* __restrict__ whh,   // [768][256] bf16 (padded)
                    const float* __restrict__ bhh,    // [762]
                    const float* __restrict__ h0,     // [64][254] or nullptr
                    int steps,
                    float* __restrict__ ctx_out,      // [64][254] or nullptr
                    __bf16* __restrict__ feat) {      // [3136][640] or nullptr
  __shared__ __bf16 h_s[RB * LDH];   //  8.25 KB
  __shared__ float  gh_s[RB * LDG];  // 48 KB
  const int tid = threadIdx.x;
  const int lane = tid & 31;
  const int wid = tid >> 5;          // 0..15
  const int b0 = blockIdx.x * RB;

  for (int i = tid; i < RB * LDH; i += 512) h_s[i] = (__bf16)0.0f;
  __syncthreads();
  if (h0) {
    for (int i = tid; i < RB * HID; i += 512) {
      int bl = i / HID, j = i - bl * HID;
      h_s[bl * LDH + j] = (__bf16)h0[(size_t)(b0 + bl) * HID + j];
    }
  }
  __syncthreads();

  for (int s = 0; s < steps; ++s) {
    // --- phase 1: gh = h @ W_hh^T + b_hh  (WMMA, 48 N-tiles over 16 waves) ---
    const __bf16* arow = h_s + (size_t)(lane & 15) * LDH;
    v16bf afrag[8];
#pragma unroll
    for (int k8 = 0; k8 < 8; ++k8) afrag[k8] = frag_ld(arow, k8 * 32, lane);

#pragma unroll 1
    for (int q = 0; q < 3; ++q) {
      const int nt = q * 16 + wid;                       // 0..47
      const __bf16* brow = whh + (size_t)(nt * 16 + (lane & 15)) * KPAD;
      v8f acc = {};
#pragma unroll
      for (int k8 = 0; k8 < 8; ++k8)
        acc = WMMA_BF16(afrag[k8], frag_ld(brow, k8 * 32, lane), acc);
      const int n = nt * 16 + (lane & 15);
      const int m0 = (lane >> 4) ? 8 : 0;
      const float bias = (n < G3) ? bhh[n] : 0.0f;
#pragma unroll
      for (int i = 0; i < 8; ++i) gh_s[(m0 + i) * LDG + n] = acc[i] + bias;
    }
    __syncthreads();

    // --- phase 2: gates (fp32 VALU) ---
    const float* gis = gi + (size_t)s * (BAT * G3);
    for (int i = tid; i < RB * HID; i += 512) {
      int bl = i / HID, j = i - bl * HID;
      int b = b0 + bl;
      float ir = gis[b * G3 + j];
      float iz = gis[b * G3 + HID + j];
      float in_ = gis[b * G3 + 2 * HID + j];
      float hr = gh_s[bl * LDG + j];
      float hz = gh_s[bl * LDG + HID + j];
      float hn = gh_s[bl * LDG + 2 * HID + j];
      float hprev = (float)h_s[bl * LDH + j];
      float r = 1.0f / (1.0f + expf(-(ir + hr)));
      float z = 1.0f / (1.0f + expf(-(iz + hz)));
      float nn = tanhf(in_ + r * hn);
      float hnew = (1.0f - z) * nn + z * hprev;
      __bf16 hb = (__bf16)hnew;
      h_s[bl * LDH + j] = hb;
      if (feat) feat[(size_t)(s * BAT + b) * FK + j] = hb;
    }
    __syncthreads();
  }

  if (ctx_out) {
    for (int i = tid; i < RB * HID; i += 512) {
      int bl = i / HID, j = i - bl * HID;
      ctx_out[(size_t)(b0 + bl) * HID + j] = (float)h_s[bl * LDH + j];
    }
  }
}

// feat[m][254..507]=context[b], feat[m][508..635]=dec_emb[m], rest 0 (h2 filled later)
__global__ void feat_fill_kernel(const float* __restrict__ ctx, const float* __restrict__ demb,
                                 __bf16* __restrict__ feat) {
  int idx = blockIdx.x * blockDim.x + threadIdx.x;
  if (idx >= (49 * BAT) * FK) return;
  int m = idx / FK, j = idx - m * FK;
  int b = m & 63;
  float v = 0.0f;
  if (j >= HID && j < 2 * HID) v = ctx[(size_t)b * HID + (j - HID)];
  else if (j >= 2 * HID && j < 2 * HID + EMB) v = demb[(size_t)m * EMB + (j - 2 * HID)];
  feat[idx] = (__bf16)v;
}

// ----------------- output projection: feat @ W_out^T + b -------------------
// 3136 x 32000, K=640. 8 waves/block; each wave -> 16(M) x 64(N) via 4 WMMA accs.
__global__ __launch_bounds__(256)
void out_gemm_kernel(const __bf16* __restrict__ feat,   // [3136][640]
                     const __bf16* __restrict__ wout,   // [32000][640]
                     const float* __restrict__ bout,    // [32000]
                     float* __restrict__ out) {         // [50*64*32000]
  const int lane = threadIdx.x & 31;
  const int wid = threadIdx.x >> 5;
  const int id = blockIdx.x * 8 + wid;   // 0..97999
  const int mt = id / 500;               // 0..195
  const int nt = id - mt * 500;          // 0..499 (64-wide N tiles)

  const __bf16* arow = feat + (size_t)(mt * 16 + (lane & 15)) * FK;
  const __bf16* brow = wout + (size_t)(nt * 64 + (lane & 15)) * FK;

  v8f acc[4] = {};
  for (int kb = 0; kb < FK; kb += 32) {
    v16bf a = frag_ld(arow, kb, lane);
    v16bf b0 = frag_ld(brow, kb, lane);
    v16bf b1 = frag_ld(brow + 16 * FK, kb, lane);
    v16bf b2 = frag_ld(brow + 32 * FK, kb, lane);
    v16bf b3 = frag_ld(brow + 48 * FK, kb, lane);
    acc[0] = WMMA_BF16(a, b0, acc[0]);
    acc[1] = WMMA_BF16(a, b1, acc[1]);
    acc[2] = WMMA_BF16(a, b2, acc[2]);
    acc[3] = WMMA_BF16(a, b3, acc[3]);
  }

  const int m0 = mt * 16 + ((lane >> 4) ? 8 : 0);
  const int nb = nt * 64 + (lane & 15);
#pragma unroll
  for (int t = 0; t < 4; ++t) {
    const int n = nb + t * 16;
    const float bias = bout[n];
#pragma unroll
    for (int i = 0; i < 8; ++i)
      out[(size_t)(m0 + i + BAT) * VOC + n] = acc[t][i] + bias;   // shifted by one step
  }
}

// ------------------------------- launcher ----------------------------------

extern "C" void kernel_launch(void* const* d_in, const int* in_sizes, int n_in,
                              void* d_out, int out_size, void* d_ws, size_t ws_size,
                              hipStream_t stream) {
  const int*   src     = (const int*)  d_in[0];   // (50,64)
  const int*   trg     = (const int*)  d_in[1];   // (50,64)
  const float* emb_enc = (const float*)d_in[2];   // (32000,128)
  const float* W_ih_e  = (const float*)d_in[3];   // (762,128)
  const float* W_hh_e  = (const float*)d_in[4];   // (762,254)
  const float* b_ih_e  = (const float*)d_in[5];
  const float* b_hh_e  = (const float*)d_in[6];
  const float* emb_dec = (const float*)d_in[7];   // (32000,128)
  const float* W_ih_d  = (const float*)d_in[8];   // (762,382)
  const float* W_hh_d  = (const float*)d_in[9];   // (762,254)
  const float* b_ih_d  = (const float*)d_in[10];
  const float* b_hh_d  = (const float*)d_in[11];
  const float* W_out   = (const float*)d_in[12];  // (32000,636)
  const float* b_out   = (const float*)d_in[13];
  float* out = (float*)d_out;

  char* w = (char*)d_ws;
  size_t off = 0;
  auto alloc = [&](size_t bytes) -> char* {
    char* p = w + off;
    off = (off + bytes + 255) & ~(size_t)255;
    return p;
  };
  float*  enc_emb  = (float*) alloc((size_t)50 * BAT * EMB * 4);
  float*  dec_emb  = (float*) alloc((size_t)49 * BAT * EMB * 4);
  float*  gi_enc   = (float*) alloc((size_t)50 * BAT * G3 * 4);
  float*  gi_dec   = (float*) alloc((size_t)49 * BAT * G3 * 4);
  float*  ctx      = (float*) alloc((size_t)BAT * HID * 4);
  __bf16* whh_e_bf = (__bf16*)alloc((size_t)768 * KPAD * 2);
  __bf16* whh_d_bf = (__bf16*)alloc((size_t)768 * KPAD * 2);
  __bf16* feat_bf  = (__bf16*)alloc((size_t)49 * BAT * FK * 2);
  __bf16* wout_bf  = (__bf16*)alloc((size_t)VOC * FK * 2);

  const int TB = 256;
  auto blocks = [](long long n, int tb) { return (unsigned)((n + tb - 1) / tb); };

  // 1) zero first output timestep
  zero_f32_kernel<<<blocks((long long)BAT * VOC, TB), TB, 0, stream>>>(out, BAT * VOC);
  // 2-3) embedding gathers
  gather_kernel<<<blocks((long long)50 * BAT * EMB, TB), TB, 0, stream>>>(src, emb_enc, enc_emb, 50 * BAT);
  gather_kernel<<<blocks((long long)49 * BAT * EMB, TB), TB, 0, stream>>>(trg, emb_dec, dec_emb, 49 * BAT);
  // 4-6) bf16 weight conversions (zero-padded)
  conv_bf16_kernel<<<blocks((long long)768 * KPAD, TB), TB, 0, stream>>>(W_hh_e, whh_e_bf, G3, HID, 768, KPAD);
  conv_bf16_kernel<<<blocks((long long)768 * KPAD, TB), TB, 0, stream>>>(W_hh_d, whh_d_bf, G3, HID, 768, KPAD);
  conv_bf16_kernel<<<blocks((long long)VOC * FK, TB), TB, 0, stream>>>(W_out, wout_bf, VOC, 636, VOC, FK);
  // 7) encoder input gates (parallel over all steps)
  gi_enc_kernel<<<blocks((long long)50 * BAT * G3, TB), TB, 0, stream>>>(enc_emb, W_ih_e, b_ih_e, gi_enc, 50 * BAT);
  // 8) encoder recurrence -> context
  gru_seq_kernel<<<4, 512, 0, stream>>>(gi_enc, whh_e_bf, b_hh_e, nullptr, 50, ctx, nullptr);
  // 9) feature buffer: context + embedding parts (h2 part filled by decoder)
  feat_fill_kernel<<<blocks((long long)49 * BAT * FK, TB), TB, 0, stream>>>(ctx, dec_emb, feat_bf);
  // 10) decoder input gates (xc = [e, context], parallel over all steps)
  gi_dec_kernel<<<blocks((long long)49 * BAT * G3, TB), TB, 0, stream>>>(dec_emb, ctx, W_ih_d, b_ih_d, gi_dec);
  // 11) decoder recurrence -> h2 written into feat
  gru_seq_kernel<<<4, 512, 0, stream>>>(gi_dec, whh_d_bf, b_hh_d, ctx, 49, nullptr, feat_bf);
  // 12) output projection: 3136x32000, K=640 (bf16 WMMA, f32 accum)
  out_gemm_kernel<<<(196 * 500) / 8, 256, 0, stream>>>(feat_bf, wout_bf, b_out, out);

  (void)in_sizes; (void)n_in; (void)out_size; (void)ws_size;
}
